// block_25829933318787
// MI455X (gfx1250) — compile-verified, matrix operand order checked by reading, not run on hardware
//
#include <hip/hip_runtime.h>

#define BATCH 2
#define CCH   64
#define NTOK  9216          // 96*96
#define EPSBN 1e-5f

typedef __attribute__((ext_vector_type(16))) __bf16 v16bf;
typedef __attribute__((ext_vector_type(8)))  __bf16 v8bf;
typedef __attribute__((ext_vector_type(8)))  float  v8f;
typedef unsigned int v4u __attribute__((ext_vector_type(4)));
typedef int          v8i __attribute__((ext_vector_type(8)));
typedef int          v4i __attribute__((ext_vector_type(4)));

#define LDS_PITCH 72        // 64 bf16 + 4-DWORD TDM pad => 144B row stride
#define LDS_TILE  (64 * LDS_PITCH + 8)   // + trailing pad slack, 16B-multiple

// ---------------------------------------------------------------------------
// Fragment loader: 16-bit A-matrix layout for V_WMMA_*_16x16x32.
// Lane l (<16) holds row l; lanes 16..31 hold the same rows, other K-half.
// Elements 0..7 = K {h*8..h*8+7}, elements 8..15 = K {16+h*8..16+h*8+7}
// => two contiguous 16-byte b128 loads per lane (global or LDS).
// ---------------------------------------------------------------------------
__device__ inline v16bf load_frag16x32(const __bf16* __restrict__ base,
                                       int row_stride, int row, int half) {
  const __bf16* r = base + (size_t)row * row_stride;
  v8bf lo = *(const v8bf*)(r + half * 8);
  v8bf hi = *(const v8bf*)(r + 16 + half * 8);
  v16bf out;
#pragma unroll
  for (int i = 0; i < 8; ++i) { out[i] = lo[i]; out[8 + i] = hi[i]; }
  return out;
}

// ---------------------------------------------------------------------------
// TDM: 2-D tile load, global -> LDS, with per-row LDS padding (32dw + 4dw).
// D# built per CDNA5 ISA 8.3/8.4 (group0 128b, group1 256b; groups 2/3 zero).
// This toolchain uses the 6-arg builtin (extra int32x8 group before cpol).
// ---------------------------------------------------------------------------
__device__ inline void tdm_load_2d(unsigned lds_addr, const void* gaddr,
                                   unsigned tensor_d0, unsigned tensor_d1,
                                   unsigned long long stride0,
                                   unsigned tile_d0, unsigned tile_d1) {
  unsigned long long ga = (unsigned long long)(uintptr_t)gaddr;
  v4u g0;
  g0[0] = 1u;                                             // count=1 (user D#)
  g0[1] = lds_addr;                                       // lds_addr[31:0]
  g0[2] = (unsigned)(ga & 0xffffffffu);                   // global_addr[31:0]
  g0[3] = (unsigned)((ga >> 32) & 0x01ffffffu) | (2u << 30); // addr[56:32]|type=2
  v8i g1;
  g1[0] = (int)((1u << 16)      // data_size = 2 bytes
              | (1u << 20)      // pad_enable
              | (4u << 22)      // pad_interval: 32 DWORDs (one 64-elem row)
              | (3u << 25));    // pad_amount: 4 DWORDs (16B)
  g1[1] = (int)((tensor_d0 & 0xffffu) << 16);             // dim0[15:0] @ bit48
  g1[2] = (int)(((tensor_d0 >> 16) & 0xffffu) | ((tensor_d1 & 0xffffu) << 16));
  g1[3] = (int)(((tensor_d1 >> 16) & 0xffffu) | ((tile_d0 & 0xffffu) << 16));
  g1[4] = (int)(tile_d1 & 0xffffu);                       // tile_dim2 = 0
  g1[5] = (int)(unsigned)(stride0 & 0xffffffffu);         // dim0_stride lo
  g1[6] = (int)(unsigned)((stride0 >> 32) & 0xffffu);     // dim0_stride hi
  g1[7] = 0;
  v4i z4 = {0, 0, 0, 0};
  v8i z8 = {0, 0, 0, 0, 0, 0, 0, 0};
  __builtin_amdgcn_tensor_load_to_lds(g0, g1, z4, z4, z8, 0);
}

// ---------------------------------------------------------------------------
// Kernel 1: 1x1-conv projections theta/phi/g, fp32 accumulate -> bf16 store.
//   theta : [B][N][C]   phi : [B][N][C]   g : [B][C][N]
// ---------------------------------------------------------------------------
__global__ __launch_bounds__(256)
void proj_kernel(const float* __restrict__ x,
                 const float* __restrict__ gw, const float* __restrict__ gb,
                 const float* __restrict__ tw, const float* __restrict__ tb,
                 const float* __restrict__ pw, const float* __restrict__ pb,
                 __bf16* __restrict__ gout, __bf16* __restrict__ tout,
                 __bf16* __restrict__ pout) {
  int idx = blockIdx.x * blockDim.x + threadIdx.x;
  if (idx >= 3 * BATCH * NTOK) return;
  const int which = idx / (BATCH * NTOK);
  const int rem   = idx % (BATCH * NTOK);
  const int b     = rem / NTOK;
  const int n     = rem % NTOK;

  const float* w    = (which == 0) ? gw : (which == 1) ? tw : pw;
  const float* bias = (which == 0) ? gb : (which == 1) ? tb : pb;

  float xv[CCH];
#pragma unroll
  for (int c = 0; c < CCH; ++c)
    xv[c] = x[((size_t)b * CCH + c) * NTOK + n];

  for (int o = 0; o < CCH; ++o) {
    float acc = bias[o];
    const float* wr = w + o * CCH;
#pragma unroll
    for (int c = 0; c < CCH; ++c) acc = fmaf(wr[c], xv[c], acc);
    __bf16 v = (__bf16)acc;
    if (which == 0)      gout[((size_t)b * CCH + o) * NTOK + n] = v;
    else if (which == 1) tout[((size_t)b * NTOK + n) * CCH + o] = v;
    else                 pout[((size_t)b * NTOK + n) * CCH + o] = v;
  }
}

// ---------------------------------------------------------------------------
// Kernel 2: flash attention, transposed-S formulation + TDM-staged tiles.
// Block = 4 waves x 16 query rows; keys stream in tiles of 64.
// Wave 0 drives a double-buffered tensor_load_to_lds pipeline (K & G tiles,
// shared by all 4 waves); S^T = K @ Q^T so softmax is lane-local and exp(S^T)
// is directly the A-fragment of P for O += P @ G.
// ---------------------------------------------------------------------------
__global__ __launch_bounds__(128)
void attn_kernel(const __bf16* __restrict__ theta,
                 const __bf16* __restrict__ phi,
                 const __bf16* __restrict__ g,
                 float* __restrict__ y) {
  __shared__ __attribute__((aligned(16))) __bf16 sh[2][2][LDS_TILE];

  const int lane = threadIdx.x & 31;
  const int wave = threadIdx.x >> 5;
  const int half = lane >> 4;
  const int l16  = lane & 15;

  const int b     = blockIdx.x / (NTOK / 64);
  const int qtile = blockIdx.x % (NTOK / 64);
  const int qbase = qtile * 64 + wave * 16;

  const __bf16* thB = theta + (size_t)b * NTOK * CCH;
  const __bf16* phB = phi   + (size_t)b * NTOK * CCH;
  const __bf16* gB  = g     + (size_t)b * CCH * NTOK;

  // Q fragments, resident across the key loop (C halves 0..31 / 32..63).
  v16bf q0 = load_frag16x32(thB + (size_t)qbase * CCH,      CCH, l16, half);
  v16bf q1 = load_frag16x32(thB + (size_t)qbase * CCH + 32, CCH, l16, half);

  v8f o0 = {0,0,0,0,0,0,0,0}, o1 = {0,0,0,0,0,0,0,0};
  v8f o2 = {0,0,0,0,0,0,0,0}, o3 = {0,0,0,0,0,0,0,0};
  float mrun = -INFINITY;   // running max for query row l16 (this lane's row)
  float lrun = 0.0f;        // running denominator for query row l16

  // Prime the pipeline: DMA tile 0 into buffer 0.
  if (wave == 0) {
    // K tile: phi rows kb..kb+63 (x64 ch); G tile: 64 ch rows x keys kb..kb+63
    tdm_load_2d((unsigned)(uintptr_t)&sh[0][0][0], phB, CCH, NTOK, CCH, 64, 64);
    tdm_load_2d((unsigned)(uintptr_t)&sh[0][1][0], gB, NTOK, CCH, NTOK, 64, 64);
  }

  for (int kb = 0; kb < NTOK; kb += 64) {
    const int buf = (kb >> 6) & 1;
    if (wave == 0) __builtin_amdgcn_s_wait_tensorcnt((short)0);
    __syncthreads();   // tile kb ready; all readers of buf^1 (iter kb-64) done
    if (wave == 0 && kb + 64 < NTOK) {
      tdm_load_2d((unsigned)(uintptr_t)&sh[buf ^ 1][0][0],
                  phB + (size_t)(kb + 64) * CCH, CCH, NTOK - (kb + 64), CCH, 64, 64);
      tdm_load_2d((unsigned)(uintptr_t)&sh[buf ^ 1][1][0],
                  gB + (kb + 64), NTOK - (kb + 64), CCH, NTOK, 64, 64);
    }
    const __bf16* kt = &sh[buf][0][0];   // [64 key rows][72] (ch contiguous)
    const __bf16* gt = &sh[buf][1][0];   // [64 ch rows][72] (keys contiguous)

    // ---- St blocks: rows = keys (VGPR idx), cols = queries (lanes) ----
    v8f st0 = {0,0,0,0,0,0,0,0}, st1 = {0,0,0,0,0,0,0,0};
    v8f st2 = {0,0,0,0,0,0,0,0}, st3 = {0,0,0,0,0,0,0,0};
    {
      v16bf ka, kb2;
      ka  = load_frag16x32(kt + 0 * 16 * LDS_PITCH,      LDS_PITCH, l16, half);
      kb2 = load_frag16x32(kt + 0 * 16 * LDS_PITCH + 32, LDS_PITCH, l16, half);
      st0 = __builtin_amdgcn_wmma_f32_16x16x32_bf16(false, ka,  false, q0, (short)0, st0, false, false);
      st0 = __builtin_amdgcn_wmma_f32_16x16x32_bf16(false, kb2, false, q1, (short)0, st0, false, false);
      ka  = load_frag16x32(kt + 1 * 16 * LDS_PITCH,      LDS_PITCH, l16, half);
      kb2 = load_frag16x32(kt + 1 * 16 * LDS_PITCH + 32, LDS_PITCH, l16, half);
      st1 = __builtin_amdgcn_wmma_f32_16x16x32_bf16(false, ka,  false, q0, (short)0, st1, false, false);
      st1 = __builtin_amdgcn_wmma_f32_16x16x32_bf16(false, kb2, false, q1, (short)0, st1, false, false);
      ka  = load_frag16x32(kt + 2 * 16 * LDS_PITCH,      LDS_PITCH, l16, half);
      kb2 = load_frag16x32(kt + 2 * 16 * LDS_PITCH + 32, LDS_PITCH, l16, half);
      st2 = __builtin_amdgcn_wmma_f32_16x16x32_bf16(false, ka,  false, q0, (short)0, st2, false, false);
      st2 = __builtin_amdgcn_wmma_f32_16x16x32_bf16(false, kb2, false, q1, (short)0, st2, false, false);
      ka  = load_frag16x32(kt + 3 * 16 * LDS_PITCH,      LDS_PITCH, l16, half);
      kb2 = load_frag16x32(kt + 3 * 16 * LDS_PITCH + 32, LDS_PITCH, l16, half);
      st3 = __builtin_amdgcn_wmma_f32_16x16x32_bf16(false, ka,  false, q0, (short)0, st3, false, false);
      st3 = __builtin_amdgcn_wmma_f32_16x16x32_bf16(false, kb2, false, q1, (short)0, st3, false, false);
    }

    // ---- lane-local row max over this lane's 32 S values ----
    float mloc = st0[0];
#pragma unroll
    for (int e = 0; e < 8; ++e) {
      mloc = fmaxf(mloc, st0[e]);
      mloc = fmaxf(mloc, st1[e]);
      mloc = fmaxf(mloc, st2[e]);
      mloc = fmaxf(mloc, st3[e]);
    }
    mloc = fmaxf(mloc, __shfl_xor(mloc, 16, 32));   // combine key halves
    float mnew  = fmaxf(mrun, mloc);
    float alpha = __expf(mrun - mnew);
    mrun = mnew;

    // ---- exp + pack: St regs ARE the P A-fragment after elementwise cvt ----
    v16bf pf0, pf1;
    float sloc = 0.0f;
#pragma unroll
    for (int e = 0; e < 8; ++e) {
      float p;
      p = __expf(st0[e] - mnew); sloc += p; pf0[e]     = (__bf16)p;
      p = __expf(st1[e] - mnew); sloc += p; pf0[8 + e] = (__bf16)p;
      p = __expf(st2[e] - mnew); sloc += p; pf1[e]     = (__bf16)p;
      p = __expf(st3[e] - mnew); sloc += p; pf1[8 + e] = (__bf16)p;
    }
    sloc += __shfl_xor(sloc, 16, 32);
    lrun = lrun * alpha + sloc;

    // ---- rescale O (rows live at v+8*half; fetch that row's alpha) ----
#pragma unroll
    for (int v = 0; v < 8; ++v) {
      float a = __shfl(alpha, v + 8 * half, 32);
      o0[v] *= a; o1[v] *= a; o2[v] *= a; o3[v] *= a;
    }

    // ---- O += P @ G (contraction = 64 keys = 2 WMMAs per 16-col block) ----
    v16bf ga, gb2;
    ga  = load_frag16x32(gt + 0 * 16 * LDS_PITCH,      LDS_PITCH, l16, half);
    gb2 = load_frag16x32(gt + 0 * 16 * LDS_PITCH + 32, LDS_PITCH, l16, half);
    o0 = __builtin_amdgcn_wmma_f32_16x16x32_bf16(false, pf0, false, ga,  (short)0, o0, false, false);
    o0 = __builtin_amdgcn_wmma_f32_16x16x32_bf16(false, pf1, false, gb2, (short)0, o0, false, false);
    ga  = load_frag16x32(gt + 1 * 16 * LDS_PITCH,      LDS_PITCH, l16, half);
    gb2 = load_frag16x32(gt + 1 * 16 * LDS_PITCH + 32, LDS_PITCH, l16, half);
    o1 = __builtin_amdgcn_wmma_f32_16x16x32_bf16(false, pf0, false, ga,  (short)0, o1, false, false);
    o1 = __builtin_amdgcn_wmma_f32_16x16x32_bf16(false, pf1, false, gb2, (short)0, o1, false, false);
    ga  = load_frag16x32(gt + 2 * 16 * LDS_PITCH,      LDS_PITCH, l16, half);
    gb2 = load_frag16x32(gt + 2 * 16 * LDS_PITCH + 32, LDS_PITCH, l16, half);
    o2 = __builtin_amdgcn_wmma_f32_16x16x32_bf16(false, pf0, false, ga,  (short)0, o2, false, false);
    o2 = __builtin_amdgcn_wmma_f32_16x16x32_bf16(false, pf1, false, gb2, (short)0, o2, false, false);
    ga  = load_frag16x32(gt + 3 * 16 * LDS_PITCH,      LDS_PITCH, l16, half);
    gb2 = load_frag16x32(gt + 3 * 16 * LDS_PITCH + 32, LDS_PITCH, l16, half);
    o3 = __builtin_amdgcn_wmma_f32_16x16x32_bf16(false, pf0, false, ga,  (short)0, o3, false, false);
    o3 = __builtin_amdgcn_wmma_f32_16x16x32_bf16(false, pf1, false, gb2, (short)0, o3, false, false);
  }

  // ---- normalize and store y[b][n][c] (f32) ----
  float linv = 1.0f / lrun;
#pragma unroll
  for (int v = 0; v < 8; ++v) {
    float li = __shfl(linv, v + 8 * half, 32);
    size_t row = (size_t)b * NTOK + qbase + v + 8 * half;
    y[row * CCH +  0 + l16] = o0[v] * li;
    y[row * CCH + 16 + l16] = o1[v] * li;
    y[row * CCH + 32 + l16] = o2[v] * li;
    y[row * CCH + 48 + l16] = o3[v] * li;
  }
}

// ---------------------------------------------------------------------------
// Kernel 3: W 1x1 conv + BatchNorm (running stats) + residual.
// ---------------------------------------------------------------------------
__global__ __launch_bounds__(256)
void out_kernel(const float* __restrict__ y,
                const float* __restrict__ ww, const float* __restrict__ wb,
                const float* __restrict__ gamma, const float* __restrict__ beta,
                const float* __restrict__ mean, const float* __restrict__ var,
                const float* __restrict__ x, float* __restrict__ out) {
  int idx = blockIdx.x * blockDim.x + threadIdx.x;
  if (idx >= BATCH * CCH * NTOK) return;
  const int n = idx % NTOK;
  const int o = (idx / NTOK) % CCH;
  const int b = idx / (CCH * NTOK);

  const float* yr = y + ((size_t)b * NTOK + n) * CCH;
  const float* wr = ww + o * CCH;
  float acc = wb[o];
#pragma unroll
  for (int c = 0; c < CCH; ++c) acc = fmaf(wr[c], yr[c], acc);
  float scale = gamma[o] * rsqrtf(var[o] + EPSBN);
  out[idx] = (acc - mean[o]) * scale + beta[o] + x[idx];
}

// ---------------------------------------------------------------------------
extern "C" void kernel_launch(void* const* d_in, const int* in_sizes, int n_in,
                              void* d_out, int out_size, void* d_ws, size_t ws_size,
                              hipStream_t stream) {
  const float* x        = (const float*)d_in[0];
  const float* g_w      = (const float*)d_in[1];
  const float* g_b      = (const float*)d_in[2];
  const float* theta_w  = (const float*)d_in[3];
  const float* theta_b  = (const float*)d_in[4];
  const float* phi_w    = (const float*)d_in[5];
  const float* phi_b    = (const float*)d_in[6];
  const float* w_w      = (const float*)d_in[7];
  const float* w_b      = (const float*)d_in[8];
  const float* bn_gamma = (const float*)d_in[9];
  const float* bn_beta  = (const float*)d_in[10];
  const float* bn_mean  = (const float*)d_in[11];
  const float* bn_var   = (const float*)d_in[12];
  float* out = (float*)d_out;

  const size_t szBF = (size_t)BATCH * NTOK * CCH * sizeof(__bf16);  // 2.36 MB
  char* ws = (char*)d_ws;
  __bf16* thetaB = (__bf16*)(ws);
  __bf16* phiB   = (__bf16*)(ws + szBF);
  __bf16* gBuf   = (__bf16*)(ws + 2 * szBF);
  float*  yBuf   = (float*)(ws + 3 * szBF);

  {
    int threads = 3 * BATCH * NTOK;
    proj_kernel<<<(threads + 255) / 256, 256, 0, stream>>>(
        x, g_w, g_b, theta_w, theta_b, phi_w, phi_b, gBuf, thetaB, phiB);
  }
  {
    int blocks = BATCH * (NTOK / 64);   // 288 blocks x 4 waves
    attn_kernel<<<blocks, 128, 0, stream>>>(thetaB, phiB, gBuf, yBuf);
  }
  {
    int threads = BATCH * CCH * NTOK;
    out_kernel<<<(threads + 255) / 256, 256, 0, stream>>>(
        yBuf, w_w, w_b, bn_gamma, bn_beta, bn_mean, bn_var, x, out);
  }
}
